// LocalAttentionParallel_5239860101839
// MI455X (gfx1250) — matrix operand order repaired and checked by reading, not compiled, not run on hardware
//
#include <hip/hip_runtime.h>
#include <hip/hip_bf16.h>

// ---------------------------------------------------------------------------
// LocalAttentionParallel on MI455X (gfx1250, wave32, WMMA)
// out = LayerNorm( band_mask(Q K^T / 512) V ),  QKV = x @ W + b
// B=4, T=2048, C=1024, span=256.
// fp32 accuracy via 2-term bf16 split (hi+lo planes, pre-packed in global):
//   a*b ~= ah*bh + ah*bl + al*bh  (fp32 accum)
// GEMM inner loops: ds_load_b128 fragments + v_wmma only.
// Staging: GLOBAL_LOAD_ASYNC_TO_LDS_B128 (ASYNCcnt) when available.
// ---------------------------------------------------------------------------

#define BB   4
#define TT   2048
#define CC   1024
#define SPAN 256

typedef __attribute__((ext_vector_type(16))) __bf16 v16bf;
typedef __attribute__((ext_vector_type(8)))  __bf16 v8bf;
typedef __attribute__((ext_vector_type(8)))  float  v8f;
typedef int v4i_vs __attribute__((vector_size(16)));   // matches builtin param

#if defined(__has_builtin)
#  if __has_builtin(__builtin_amdgcn_global_load_async_to_lds_b128)
#    define USE_ASYNC_LDS 1
#  endif
#endif

// 16-byte global -> LDS copy. Async DMA path (ASYNCcnt) when the toolchain
// exposes it; falls back to load+store. Generic->AS casts via uintptr:
// generic LDS ptr low 32 bits are the LDS offset (flat aperture rule).
static __device__ inline void cp16(void* lds, const void* g) {
#ifdef USE_ASYNC_LDS
  __builtin_amdgcn_global_load_async_to_lds_b128(
      (__attribute__((address_space(1))) v4i_vs*)(uintptr_t)g,
      (__attribute__((address_space(3))) v4i_vs*)(unsigned int)(uintptr_t)lds,
      0, 0);
#else
  *(uint4*)lds = *(const uint4*)g;
#endif
}

static __device__ inline void st16_zero(void* lds) {
  uint4 z;
  z.x = z.y = z.z = z.w = 0u;
  *(uint4*)lds = z;
}

// Wait for this wave's async staging, then workgroup barrier.
static __device__ inline void stage_fence() {
#ifdef USE_ASYNC_LDS
#  if __has_builtin(__builtin_amdgcn_s_wait_asynccnt)
  __builtin_amdgcn_s_wait_asynccnt(0);
#  else
  asm volatile("s_wait_asynccnt 0x0" ::: "memory");
#  endif
#endif
  __syncthreads();
}

static __device__ inline void split_bf16(float x, __bf16& h, __bf16& l) {
  h = (__bf16)x;
  l = (__bf16)(x - (float)h);
}

// Build a 16-element bf16 fragment from two 16-byte LDS loads.
static __device__ inline v16bf ldsfrag(const __bf16* base, int o0, int o1) {
  v8bf a = *(const v8bf*)(base + o0);
  v8bf b = *(const v8bf*)(base + o1);
  return __builtin_shufflevector(a, b, 0, 1, 2, 3, 4, 5, 6, 7,
                                       8, 9, 10, 11, 12, 13, 14, 15);
}

// A-matrix 16x32 fragment. LDS tile row-major [row][k], stride ld (bf16 elems).
// lane L<16: M=L, K={0..7,16..23}; L>=16: M=L-16, K={8..15,24..31}
static __device__ inline v16bf frag_a(const __bf16* tile, int ld, int lane) {
  const int row  = lane & 15;
  const int koff = (lane < 16) ? 0 : 8;
  return ldsfrag(tile + row * ld, koff, koff + 16);
}

// B-matrix 32x16 fragment. LDS tile TRANSPOSED [n][k], stride ld.
// lane L<16: N=L, K={0..7,16..23}; L>=16: N=L-16, K={8..15,24..31}
static __device__ inline v16bf frag_b(const __bf16* tile, int ld, int lane) {
  const int n    = lane & 15;
  const int koff = (lane < 16) ? 0 : 8;
  return ldsfrag(tile + n * ld, koff, koff + 16);
}

static __device__ inline v8f wmma3(v16bf ah, v16bf al, v16bf bh, v16bf bl, v8f c) {
  c = __builtin_amdgcn_wmma_f32_16x16x32_bf16(false, ah, false, bh, (short)0, c, false, false);
  c = __builtin_amdgcn_wmma_f32_16x16x32_bf16(false, ah, false, bl, (short)0, c, false, false);
  c = __builtin_amdgcn_wmma_f32_16x16x32_bf16(false, al, false, bh, (short)0, c, false, false);
  return c;
}

// ---------------------------------------------------------------------------
// Pack kernels (one-time): fp32 -> bf16 hi/lo planes.
// ---------------------------------------------------------------------------
__global__ __launch_bounds__(256) void pack_x_kernel(
    const float* __restrict__ x, __bf16* __restrict__ xh, __bf16* __restrict__ xl) {
  const size_t idx = ((size_t)blockIdx.x * 256 + threadIdx.x) * 8;
  float4 a = *(const float4*)&x[idx];
  float4 c = *(const float4*)&x[idx + 4];
  float f[8] = {a.x, a.y, a.z, a.w, c.x, c.y, c.z, c.w};
  v8bf h, l;
#pragma unroll
  for (int i = 0; i < 8; ++i) {
    __bf16 hh, ll;
    split_bf16(f[i], hh, ll);
    h[i] = hh; l[i] = ll;
  }
  *(v8bf*)&xh[idx] = h;
  *(v8bf*)&xl[idx] = l;
}

// W [1024 k][3072 n] fp32 -> Wt_hi/Wt_lo [3072 n][1024 k] bf16 (LDS-tiled 64x64)
__global__ __launch_bounds__(256) void pack_w_kernel(
    const float* __restrict__ W, __bf16* __restrict__ Wth, __bf16* __restrict__ Wtl) {
  __shared__ __align__(16) __bf16 th[64 * 72];
  __shared__ __align__(16) __bf16 tl[64 * 72];
  const int tid = threadIdx.x;
  const int kb = (blockIdx.x / 48) * 64;   // 1024/64 = 16 k-tiles
  const int nb = (blockIdx.x % 48) * 64;   // 3072/64 = 48 n-tiles
  for (int e = tid; e < 64 * 16; e += 256) {       // 64 k-rows x 16 float4
    int r = e >> 4, c4 = e & 15;
    float4 vv = *(const float4*)&W[(size_t)(kb + r) * (3 * CC) + nb + (c4 << 2)];
    float f[4] = {vv.x, vv.y, vv.z, vv.w};
#pragma unroll
    for (int i = 0; i < 4; ++i) {
      __bf16 h, l;
      split_bf16(f[i], h, l);
      th[(c4 * 4 + i) * 72 + r] = h;               // transpose into LDS
      tl[(c4 * 4 + i) * 72 + r] = l;
    }
  }
  __syncthreads();
  for (int e = tid; e < 64 * 8; e += 256) {        // 64 n-rows x 8 x (8 bf16)
    int n = e >> 3, g = e & 7;
    *(uint4*)&Wth[(size_t)(nb + n) * CC + kb + g * 8] = *(const uint4*)&th[n * 72 + g * 8];
    *(uint4*)&Wtl[(size_t)(nb + n) * CC + kb + g * 8] = *(const uint4*)&tl[n * 72 + g * 8];
  }
}

// ---------------------------------------------------------------------------
// Kernel 1: qkv = x @ W + b  (M=8192, N=3072, K=1024), block tile 64x128.
// Outputs: q,k as [t][c] hi/lo planes; v TRANSPOSED [b][c][t] hi/lo planes.
// ---------------------------------------------------------------------------
__global__ __launch_bounds__(256) void qkv_gemm_kernel(
    const __bf16* __restrict__ xh, const __bf16* __restrict__ xl,
    const __bf16* __restrict__ Wth, const __bf16* __restrict__ Wtl,
    const float* __restrict__ bias,
    __bf16* __restrict__ qh, __bf16* __restrict__ ql,
    __bf16* __restrict__ kh, __bf16* __restrict__ kl,
    __bf16* __restrict__ vth, __bf16* __restrict__ vtl) {
  __shared__ __align__(16) __bf16 sAh[64 * 40], sAl[64 * 40];     // [m][k]
  __shared__ __align__(16) __bf16 sBh[128 * 40], sBl[128 * 40];   // [n][k]
  const int tid  = threadIdx.x;
  const int lane = tid & 31;
  const int w    = tid >> 5;
  const int nTilesN = (3 * CC) / 128;  // 24
  const int m0 = (blockIdx.x / nTilesN) * 64;
  const int n0 = (blockIdx.x % nTilesN) * 128;

  v8f acc[4] = {v8f{}, v8f{}, v8f{}, v8f{}};

  for (int kc = 0; kc < CC; kc += 32) {
    __syncthreads();
    {  // stage A: 64 rows x 32 bf16, one 16B group per thread per plane
      int r = tid >> 2, g = (tid & 3) * 8;
      cp16(&sAh[r * 40 + g], &xh[(size_t)(m0 + r) * CC + kc + g]);
      cp16(&sAl[r * 40 + g], &xl[(size_t)(m0 + r) * CC + kc + g]);
    }
    for (int e = tid; e < 512; e += 256) {  // stage B: 128 n-rows x 32 bf16
      int r = e >> 2, g = (e & 3) * 8;
      cp16(&sBh[r * 40 + g], &Wth[(size_t)(n0 + r) * CC + kc + g]);
      cp16(&sBl[r * 40 + g], &Wtl[(size_t)(n0 + r) * CC + kc + g]);
    }
    if (kc + 32 < CC)
      __builtin_prefetch(xh + (size_t)(m0 + (tid >> 2)) * CC + kc + 32, 0, 1);
    stage_fence();

    v16bf bh = frag_b(sBh + w * 16 * 40, 40, lane);
    v16bf bl = frag_b(sBl + w * 16 * 40, 40, lane);
#pragma unroll
    for (int i = 0; i < 4; ++i) {
      v16bf ah = frag_a(sAh + i * 16 * 40, 40, lane);
      v16bf al = frag_a(sAl + i * 16 * 40, 40, lane);
      acc[i] = wmma3(ah, al, bh, bl, acc[i]);
    }
  }

  const int col = lane & 15;
  const int rowoff = (lane < 16) ? 0 : 8;
#pragma unroll
  for (int i = 0; i < 4; ++i) {
#pragma unroll
    for (int r = 0; r < 8; ++r) {
      int m = m0 + i * 16 + rowoff + r;
      int n = n0 + w * 16 + col;
      float val = acc[i][r] + bias[n];
      __bf16 h, l;
      split_bf16(val, h, l);
      int sel = n >> 10, c = n & 1023;
      if (sel == 0) {
        size_t o = (size_t)m * CC + c;
        qh[o] = h; ql[o] = l;
      } else if (sel == 1) {
        size_t o = (size_t)m * CC + c;
        kh[o] = h; kl[o] = l;
      } else {  // v transposed: [b][c][t]
        int b = m >> 11, t = m & (TT - 1);
        size_t o = ((size_t)b * CC + c) * TT + t;
        vth[o] = h; vtl[o] = l;
      }
    }
  }
}

// ---------------------------------------------------------------------------
// Kernel 2: banded scores S[b,blk,il,u] = mask * (q_i . k_j) / 512
//   u in [0,320), j = blk*64 - 256 + u. Output: bf16 hi/lo planes.
// K rows [j][c] already serve as the transposed-B [n][k] layout.
// ---------------------------------------------------------------------------
__global__ __launch_bounds__(256) void scores_kernel(
    const __bf16* __restrict__ qh, const __bf16* __restrict__ ql,
    const __bf16* __restrict__ kh, const __bf16* __restrict__ kl,
    __bf16* __restrict__ Sh, __bf16* __restrict__ Sl) {
  __shared__ __align__(16) __bf16 sQh[64 * 40], sQl[64 * 40];     // [i][c]
  __shared__ __align__(16) __bf16 sKh[320 * 40], sKl[320 * 40];   // [j][c]
  const int tid  = threadIdx.x;
  const int lane = tid & 31;
  const int w    = tid >> 5;
  const int blk  = blockIdx.x & 31;
  const int b    = blockIdx.x >> 5;
  const int i0   = blk * 64;
  const long jstart = (long)i0 - SPAN;

  const __bf16* qhrow = qh + ((size_t)b * TT + i0) * CC;
  const __bf16* qlrow = ql + ((size_t)b * TT + i0) * CC;
  const __bf16* khbase = kh + (size_t)b * TT * CC;
  const __bf16* klbase = kl + (size_t)b * TT * CC;

  v8f acc[10];
#pragma unroll
  for (int t = 0; t < 10; ++t) acc[t] = v8f{};

  for (int kc = 0; kc < CC; kc += 32) {
    __syncthreads();
    {  // stage Q: 64 x 32
      int r = tid >> 2, g = (tid & 3) * 8;
      cp16(&sQh[r * 40 + g], &qhrow[(size_t)r * CC + kc + g]);
      cp16(&sQl[r * 40 + g], &qlrow[(size_t)r * CC + kc + g]);
    }
    for (int e = tid; e < 1280; e += 256) {  // stage K: 320 x 32, zero j<0
      int r = e >> 2, g = (e & 3) * 8;
      long j = jstart + r;
      if (j >= 0) {
        cp16(&sKh[r * 40 + g], &khbase[(size_t)j * CC + kc + g]);
        cp16(&sKl[r * 40 + g], &klbase[(size_t)j * CC + kc + g]);
      } else {
        st16_zero(&sKh[r * 40 + g]);
        st16_zero(&sKl[r * 40 + g]);
      }
    }
    stage_fence();

#pragma unroll
    for (int t = 0; t < 10; ++t) {
      int job  = w + 8 * t;      // 80 tile-jobs: 4 i-sub x 20 j-tiles
      int isub = job / 20;
      int jt   = job % 20;
      v16bf ah = frag_a(sQh + isub * 16 * 40, 40, lane);
      v16bf al = frag_a(sQl + isub * 16 * 40, 40, lane);
      v16bf bh = frag_b(sKh + jt * 16 * 40, 40, lane);
      v16bf bl = frag_b(sKl + jt * 16 * 40, 40, lane);
      acc[t] = wmma3(ah, al, bh, bl, acc[t]);
    }
  }

  const float inv_scale = 1.0f / 512.0f;    // 1/sqrt(C*span)
  const int col = lane & 15;
  const int rowoff = (lane < 16) ? 0 : 8;
  const size_t sbase = (((size_t)b * 32 + blk) * 64) * 320;
#pragma unroll
  for (int t = 0; t < 10; ++t) {
    int job  = w + 8 * t;
    int isub = job / 20;
    int jt   = job % 20;
#pragma unroll
    for (int r = 0; r < 8; ++r) {
      int il = isub * 16 + rowoff + r;
      long i = i0 + il;
      int  u = jt * 16 + col;
      long j = jstart + u;
      float m = (j >= 0 && j <= i && j > i - SPAN) ? inv_scale : 0.0f;
      float val = acc[t][r] * m;
      __bf16 h, l;
      split_bf16(val, h, l);
      size_t o = sbase + (size_t)il * 320 + u;
      Sh[o] = h; Sl[o] = l;
    }
  }
}

// ---------------------------------------------------------------------------
// Kernel 3: out[64 x 1024] = S[64 x 320] @ V[320 x 1024] per (b, blk, nchunk).
// vt planes [b][c][t] serve as the transposed-B [n][k] layout directly.
// ---------------------------------------------------------------------------
__global__ __launch_bounds__(256) void av_kernel(
    const __bf16* __restrict__ Sh, const __bf16* __restrict__ Sl,
    const __bf16* __restrict__ vth, const __bf16* __restrict__ vtl,
    float* __restrict__ out) {
  __shared__ __align__(16) __bf16 sSh[64 * 40], sSl[64 * 40];     // [i][u]
  __shared__ __align__(16) __bf16 sVh[128 * 40], sVl[128 * 40];   // [c][u]
  const int tid  = threadIdx.x;
  const int lane = tid & 31;
  const int w    = tid >> 5;
  const int nch  = blockIdx.x & 7;
  const int blk  = (blockIdx.x >> 3) & 31;
  const int b    = blockIdx.x >> 8;
  const int i0   = blk * 64;
  const int n0   = nch * 128;
  const long jstart = (long)i0 - SPAN;

  const __bf16* Shb = Sh + (((size_t)b * 32 + blk) * 64) * 320;
  const __bf16* Slb = Sl + (((size_t)b * 32 + blk) * 64) * 320;
  const __bf16* vhb = vth + (size_t)b * CC * TT;
  const __bf16* vlb = vtl + (size_t)b * CC * TT;

  v8f acc[4] = {v8f{}, v8f{}, v8f{}, v8f{}};

  for (int u = 0; u < 320; u += 32) {
    __syncthreads();
    {  // stage S: 64 x 32
      int r = tid >> 2, g = (tid & 3) * 8;
      cp16(&sSh[r * 40 + g], &Shb[(size_t)r * 320 + u + g]);
      cp16(&sSl[r * 40 + g], &Slb[(size_t)r * 320 + u + g]);
    }
    for (int e = tid; e < 512; e += 256) {  // stage Vt: 128 c-rows x 32 t
      int r = e >> 2, g = (e & 3) * 8;
      long j = jstart + u + g;              // 8-elem group wholly valid or not
      if (j >= 0) {
        cp16(&sVh[r * 40 + g], &vhb[(size_t)(n0 + r) * TT + j]);
        cp16(&sVl[r * 40 + g], &vlb[(size_t)(n0 + r) * TT + j]);
      } else {
        st16_zero(&sVh[r * 40 + g]);
        st16_zero(&sVl[r * 40 + g]);
      }
    }
    stage_fence();

    v16bf bh = frag_b(sVh + w * 16 * 40, 40, lane);
    v16bf bl = frag_b(sVl + w * 16 * 40, 40, lane);
#pragma unroll
    for (int i = 0; i < 4; ++i) {
      v16bf ah = frag_a(sSh + i * 16 * 40, 40, lane);
      v16bf al = frag_a(sSl + i * 16 * 40, 40, lane);
      acc[i] = wmma3(ah, al, bh, bl, acc[i]);
    }
  }

  const int col = lane & 15;
  const int rowoff = (lane < 16) ? 0 : 8;
#pragma unroll
  for (int i = 0; i < 4; ++i) {
#pragma unroll
    for (int r = 0; r < 8; ++r) {
      int m = i0 + i * 16 + rowoff + r;
      int n = n0 + w * 16 + col;
      out[((size_t)b * TT + m) * CC + n] = acc[i][r];
    }
  }
}

// ---------------------------------------------------------------------------
// Kernel 4: in-place LayerNorm over C=1024 per row of d_out.
// ---------------------------------------------------------------------------
__global__ __launch_bounds__(256) void ln_kernel(
    float* __restrict__ out, const float* __restrict__ lnw,
    const float* __restrict__ lnb) {
  __shared__ float s1[256];
  __shared__ float s2[256];
  const int tid = threadIdx.x;
  const size_t base = (size_t)blockIdx.x * CC + tid * 4;
  float4 xv = *(const float4*)&out[base];
  s1[tid] = xv.x + xv.y + xv.z + xv.w;
  s2[tid] = xv.x * xv.x + xv.y * xv.y + xv.z * xv.z + xv.w * xv.w;
  for (int off = 128; off > 0; off >>= 1) {
    __syncthreads();
    if (tid < off) { s1[tid] += s1[tid + off]; s2[tid] += s2[tid + off]; }
  }
  __syncthreads();
  const float mean = s1[0] * (1.0f / CC);
  const float var  = s2[0] * (1.0f / CC) - mean * mean;
  const float rs   = rsqrtf(var + 1e-5f);
  const int c = tid * 4;
  float4 yv;
  yv.x = (xv.x - mean) * rs * lnw[c + 0] + lnb[c + 0];
  yv.y = (xv.y - mean) * rs * lnw[c + 1] + lnb[c + 1];
  yv.z = (xv.z - mean) * rs * lnw[c + 2] + lnb[c + 2];
  yv.w = (xv.w - mean) * rs * lnw[c + 3] + lnb[c + 3];
  *(float4*)&out[base] = yv;
}

// ---------------------------------------------------------------------------
extern "C" void kernel_launch(void* const* d_in, const int* in_sizes, int n_in,
                              void* d_out, int out_size, void* d_ws, size_t ws_size,
                              hipStream_t stream) {
  const float* x    = (const float*)d_in[0];
  const float* Wqkv = (const float*)d_in[1];
  const float* bqkv = (const float*)d_in[2];
  const float* lnw  = (const float*)d_in[3];
  const float* lnb  = (const float*)d_in[4];
  float* out = (float*)d_out;

  char* ws = (char*)d_ws;
  size_t off = 0;
  const size_t planeB = (size_t)BB * TT * CC * sizeof(__bf16);     // 16.8 MB
  const size_t wB     = (size_t)CC * 3 * CC * sizeof(__bf16);      // 6.3 MB
  const size_t sB     = (size_t)BB * 32 * 64 * 320 * sizeof(__bf16);

  __bf16* xh  = (__bf16*)(ws + off); off += planeB;
  __bf16* xl  = (__bf16*)(ws + off); off += planeB;
  __bf16* Wth = (__bf16*)(ws + off); off += wB;
  __bf16* Wtl = (__bf16*)(ws + off); off += wB;
  __bf16* qhp = (__bf16*)(ws + off); off += planeB;
  __bf16* qlp = (__bf16*)(ws + off); off += planeB;
  __bf16* khp = (__bf16*)(ws + off); off += planeB;
  __bf16* klp = (__bf16*)(ws + off); off += planeB;
  __bf16* vth = (__bf16*)(ws + off); off += planeB;
  __bf16* vtl = (__bf16*)(ws + off); off += planeB;
  __bf16* Shp = (__bf16*)(ws + off); off += sB;
  __bf16* Slp = (__bf16*)(ws + off); off += sB;

  // 0) pack inputs into bf16 hi/lo planes (W also transposed)
  pack_x_kernel<<<dim3((BB * TT * CC) / (256 * 8)), 256, 0, stream>>>(x, xh, xl);
  pack_w_kernel<<<dim3(16 * 48), 256, 0, stream>>>(Wqkv, Wth, Wtl);
  // 1) QKV projection
  qkv_gemm_kernel<<<dim3((BB * TT / 64) * ((3 * CC) / 128)), 256, 0, stream>>>(
      xh, xl, Wth, Wtl, bqkv, qhp, qlp, khp, klp, vth, vtl);
  // 2) banded scores
  scores_kernel<<<dim3(BB * (TT / 64)), 256, 0, stream>>>(qhp, qlp, khp, klp, Shp, Slp);
  // 3) S @ V -> d_out
  av_kernel<<<dim3(BB * (TT / 64) * (CC / 128)), 256, 0, stream>>>(Shp, Slp, vth, vtl, out);
  // 4) in-place LayerNorm
  ln_kernel<<<dim3(BB * TT), 256, 0, stream>>>(out, lnw, lnb);
}
